// TaskRelevanceQualityEstimator_60524679135567
// MI455X (gfx1250) — matrix operand order repaired and checked by reading, not compiled, not run on hardware
//
#include <hip/hip_runtime.h>
#include <hip/hip_bf16.h>
#include <math.h>

typedef __attribute__((ext_vector_type(16))) __bf16 v16bf;
typedef __attribute__((ext_vector_type(8)))  __bf16 v8bf;
typedef __attribute__((ext_vector_type(8)))  float  v8f;

#define B_ROWS 16384
#define H_DIM  768

__device__ __forceinline__ float sigm(float x) { return 1.0f / (1.0f + expf(-x)); }
__device__ __forceinline__ float gelu_exact(float x) {
    return 0.5f * x * (1.0f + erff(x * 0.70710678118654752440f));
}

// ---------------------------------------------------------------------------
// Convert + pack weights into WMMA B-fragment order:
//   Wp[(((kt*N + n)*2 + h)*16 + j] = (bf16) W[kt*32 + h*16 + j][n]
// so each lane's 16 B values (fixed n, K-half h, k-tile kt) are one
// contiguous 32-byte chunk in global memory.
// ---------------------------------------------------------------------------
__global__ __launch_bounds__(256)
void pack_w_bf16(const float* __restrict__ W, __bf16* __restrict__ Wp, int K, int N) {
    const int g = blockIdx.x * blockDim.x + threadIdx.x;
    const int total = (K >> 5) * N * 2;
    if (g >= total) return;
    const int h  = g & 1;
    const int n  = (g >> 1) % N;
    const int kt = (g >> 1) / N;
    const float* src = W + (size_t)(kt * 32 + h * 16) * N + n;
    __bf16* dst = Wp + (size_t)g * 16;
    #pragma unroll
    for (int j = 0; j < 16; ++j) dst[j] = (__bf16)src[(size_t)j * N];
}

// ---------------------------------------------------------------------------
// Tiled WMMA GEMM: Out[B x N] = act( A[B x K] * Wp[K x N, packed] + bias )
// BM=128, BN=64, BK=32; 256 threads = 8 waves; wave w owns rows [w*16,w*16+16)
// and all 64 cols -> 4 f32 accumulators.
// A fragment (16-bit A 16x32): lane&15 = M, h=lane>>4;
//   elems 0..15 = K[16h..16h+15]  (2x b128 from LDS)
// B fragment: contiguous 32B from packed Wp (2x global b128), no LDS.
// D: vgpr r -> M=(lane>>4)*8+r, N=lane&15.
// ---------------------------------------------------------------------------
template<bool A_IS_F32, bool CONCAT, bool DO_GELU, bool OUT_BF16>
__global__ __launch_bounds__(256, 2)
void gemm_bias_act(const void* __restrict__ A0, const void* __restrict__ A1,
                   int lda, int K0, int K,
                   const __bf16* __restrict__ Wp, const float* __restrict__ bias,
                   void* __restrict__ Out, int N)
{
    constexpr int BM = 128, BK = 32;
    constexpr int LDA = BK + 8;   // bf16 elements, padded (80B rows, 16B aligned)

    __shared__ __bf16 sA[BM * LDA];

    const int tid  = threadIdx.x;
    const int wave = tid >> 5;
    const int lane = tid & 31;
    const int rowBlk = blockIdx.x * BM;
    const int colBlk = blockIdx.y * 64;

    const int nf = lane & 15;          // N within 16-subtile / M within A frag
    const int h  = lane >> 4;          // K-half select
    const int kh = h * 8;

    v8f acc[4] = {};

    for (int k0 = 0; k0 < K; k0 += BK) {
        const int kt = k0 >> 5;

        // ---- B fragments straight from packed global (overlaps A staging) ----
        v16bf bfr[4];
        #pragma unroll
        for (int t = 0; t < 4; ++t) {
            const int ng = colBlk + t * 16 + nf;
            bfr[t] = *(const v16bf*)(Wp + (((size_t)kt * N + ng) * 2 + h) * 16);
        }

        // ---- stage A tile: 128x32, 2 threads per row, 16 elems each ----
        {
            const int r  = tid >> 1;
            const int c  = (tid & 1) * 16;
            const int gr = rowBlk + r;
            const int gk = k0 + c;             // 16-chunk never crosses K0 (all mult of 32)
            if (A_IS_F32) {
                const float* src; int kk;
                if (CONCAT && gk >= K0) { src = (const float*)A1; kk = gk - K0; }
                else                    { src = (const float*)A0; kk = gk; }
                const float* p = src + (size_t)gr * lda + kk;
                if (k0 + BK < K) __builtin_prefetch(p + BK, 0, 0);
                #pragma unroll
                for (int j = 0; j < 16; ++j) sA[r * LDA + c + j] = (__bf16)p[j];
            } else {
                const __bf16* p = (const __bf16*)A0 + (size_t)gr * lda + gk;
                if (k0 + BK < K) __builtin_prefetch(p + BK, 0, 0);
                #pragma unroll
                for (int j = 0; j < 16; ++j) sA[r * LDA + c + j] = p[j];
            }
        }
        __syncthreads();

        // ---- A fragment: two 16B vector loads from LDS ----
        const __bf16* pa = &sA[(wave * 16 + nf) * LDA];
        const v8bf a0 = *(const v8bf*)(pa + h * 16);
        const v8bf a1 = *(const v8bf*)(pa + h * 16 + 8);
        const v16bf a = __builtin_shufflevector(a0, a1,
            0, 1, 2, 3, 4, 5, 6, 7, 8, 9, 10, 11, 12, 13, 14, 15);

        // ---- 4 WMMAs over the 64-wide N strip ----
        #pragma unroll
        for (int t = 0; t < 4; ++t)
            acc[t] = __builtin_amdgcn_wmma_f32_16x16x32_bf16(
                false, a, false, bfr[t], (short)0, acc[t], false, false);

        __syncthreads();
    }

    // ---- epilogue: bias + (GELU) + store ----
    const int mb = wave * 16 + h * 8;
    #pragma unroll
    for (int t = 0; t < 4; ++t) {
        const int gcol = colBlk + t * 16 + nf;
        const float bv = bias[gcol];
        #pragma unroll
        for (int r = 0; r < 8; ++r) {
            float x = acc[t][r] + bv;
            if (DO_GELU) x = gelu_exact(x);
            const size_t idx = (size_t)(rowBlk + mb + r) * N + gcol;
            if (OUT_BF16) ((__bf16*)Out)[idx] = (__bf16)x;
            else          ((float*)Out)[idx]  = x;
        }
    }
}

// ---------------------------------------------------------------------------
// geometric(): wave per row. norm, sparsity, std (ddof=1). Saves norm.
// ---------------------------------------------------------------------------
__global__ __launch_bounds__(256)
void geometric_kernel(const float* __restrict__ f, float* __restrict__ outRow,
                      float* __restrict__ normOut)
{
    const int wave = threadIdx.x >> 5, lane = threadIdx.x & 31;
    const int row  = blockIdx.x * 8 + wave;
    const float* p = f + (size_t)row * H_DIM;
    float s1 = 0.f, s2 = 0.f, cnt = 0.f;
    for (int j = lane; j < H_DIM; j += 32) {
        float v = p[j];
        s1 += v; s2 += v * v;
        cnt += (fabsf(v) > 0.01f) ? 1.f : 0.f;
    }
    #pragma unroll
    for (int o = 16; o > 0; o >>= 1) {
        s1 += __shfl_xor(s1, o, 32);
        s2 += __shfl_xor(s2, o, 32);
        cnt += __shfl_xor(cnt, o, 32);
    }
    if (lane == 0) {
        float n  = sqrtf(s2);
        float nq = sigm((n - 1.0f) * 2.0f);
        float sp = cnt / (float)H_DIM;
        float var = (s2 - s1 * s1 / (float)H_DIM) / (float)(H_DIM - 1);
        float sq = sigm(sqrtf(fmaxf(var, 0.f)) * 10.0f - 1.0f);
        outRow[row]  = (nq + sp + sq) * (1.0f / 3.0f);
        normOut[row] = n;
    }
}

// ---------------------------------------------------------------------------
// information() tail: entropy of histogram of round(sigmoid(enc)*10) (11 bins)
// plus the norm-derived terms. Thread per row.
// ---------------------------------------------------------------------------
__global__ __launch_bounds__(256)
void info_finish_kernel(const float* __restrict__ enc, const float* __restrict__ norms,
                        float* __restrict__ outRow)
{
    const int row = blockIdx.x * blockDim.x + threadIdx.x;
    if (row >= B_ROWS) return;
    const float* e = enc + (size_t)row * 64;
    int hist[11];
    #pragma unroll
    for (int i = 0; i < 11; ++i) hist[i] = 0;
    for (int j = 0; j < 64; ++j) {
        float s = sigm(e[j]);
        int d = (int)rintf(s * 10.0f);           // round-half-even, matches jnp.round
        d = d < 0 ? 0 : (d > 10 ? 10 : d);
        hist[d]++;
    }
    float ent = 0.f;
    #pragma unroll
    for (int i = 0; i < 11; ++i) {
        if (hist[i] > 0) {
            float pr = (float)hist[i] * (1.0f / 64.0f);
            ent -= pr * logf(pr + 1e-8f);
        }
    }
    float eq = sigm(ent - 2.0f);
    float n  = norms[row];
    float cq = ((n > 0.01f) ? 1.0f : 0.0f) / (float)H_DIM;
    float nm = fmaxf(n, 1e-12f);
    float dq = sigm((n * n) / (nm * nm) - 0.5f);   // sum(fn*fn) analytically
    outRow[row] = (eq + cq + dq) * (1.0f / 3.0f);
}

// ---------------------------------------------------------------------------
// final layer: sigmoid( act[row,:Kd] . w + b ). Wave per row, act in bf16.
// ---------------------------------------------------------------------------
__global__ __launch_bounds__(256)
void dot_sigmoid_kernel(const __bf16* __restrict__ act, int Kd,
                        const float* __restrict__ w, const float* __restrict__ b,
                        float* __restrict__ outRow)
{
    const int wave = threadIdx.x >> 5, lane = threadIdx.x & 31;
    const int row  = blockIdx.x * 8 + wave;
    const __bf16* p = act + (size_t)row * Kd;
    float s = 0.f;
    for (int j = lane; j < Kd; j += 32) s += (float)p[j] * w[j];
    #pragma unroll
    for (int o = 16; o > 0; o >>= 1) s += __shfl_xor(s, o, 32);
    if (lane == 0) outRow[row] = sigm(s + b[0]);
}

__global__ __launch_bounds__(256)
void overall_kernel(float* __restrict__ out)
{
    const int i = blockIdx.x * blockDim.x + threadIdx.x;
    if (i >= B_ROWS) return;
    const float v = out[0 * B_ROWS + i] + out[1 * B_ROWS + i] +
                    out[4 * B_ROWS + i] + out[5 * B_ROWS + i] +
                    out[7 * B_ROWS + i] +
                    (out[8 * B_ROWS + i] + out[9 * B_ROWS + i]) * 0.5f;
    out[17 * B_ROWS + i] = v * (1.0f / 6.0f);
}

// ---------------------------------------------------------------------------
extern "C" void kernel_launch(void* const* d_in, const int* in_sizes, int n_in,
                              void* d_out, int out_size, void* d_ws, size_t ws_size,
                              hipStream_t stream)
{
    (void)in_sizes; (void)n_in; (void)out_size; (void)ws_size;

    const float* feats[4] = { (const float*)d_in[0], (const float*)d_in[1],
                              (const float*)d_in[2], (const float*)d_in[3] };
    const float* info_w1 = (const float*)d_in[5];  const float* info_b1 = (const float*)d_in[6];
    const float* info_w2 = (const float*)d_in[7];  const float* info_b2 = (const float*)d_in[8];
    const float* imp_w1  = (const float*)d_in[9];  const float* imp_b1  = (const float*)d_in[10];
    const float* imp_w2  = (const float*)d_in[11]; const float* imp_b2  = (const float*)d_in[12];
    const float* disc_w1 = (const float*)d_in[13]; const float* disc_b1 = (const float*)d_in[14];
    const float* disc_w2 = (const float*)d_in[15]; const float* disc_b2 = (const float*)d_in[16];
    const float* disc_w3 = (const float*)d_in[17]; const float* disc_b3 = (const float*)d_in[18];
    const float* cons_w1 = (const float*)d_in[19]; const float* cons_b1 = (const float*)d_in[20];
    const float* cons_w2 = (const float*)d_in[21]; const float* cons_b2 = (const float*)d_in[22];
    const float* cons_w3 = (const float*)d_in[23]; const float* cons_b3 = (const float*)d_in[24];
    const float* diff_w1 = (const float*)d_in[25]; const float* diff_b1 = (const float*)d_in[26];
    const float* diff_w2 = (const float*)d_in[27]; const float* diff_b2 = (const float*)d_in[28];
    const float* diff_w3 = (const float*)d_in[29]; const float* diff_b3 = (const float*)d_in[30];

    float* out = (float*)d_out;

    // ---- workspace carve-up (256B aligned regions) ----
    uint8_t* ws = (uint8_t*)d_ws;
    size_t off = 0;
    auto take = [&](size_t bytes) -> void* {
        void* p = ws + off;
        off += (bytes + 255) & ~(size_t)255;
        return p;
    };
    __bf16* wbInfo1 = (__bf16*)take((size_t)768 * 256 * 2);
    __bf16* wbInfo2 = (__bf16*)take((size_t)256 * 64 * 2);
    __bf16* wbImp1  = (__bf16*)take((size_t)768 * 384 * 2);
    __bf16* wbDisc1 = (__bf16*)take((size_t)768 * 256 * 2);
    __bf16* wbDisc2 = (__bf16*)take((size_t)256 * 64 * 2);
    __bf16* wbCons1 = (__bf16*)take((size_t)1536 * 768 * 2);
    __bf16* wbCons2 = (__bf16*)take((size_t)768 * 384 * 2);
    __bf16* wbDiff1 = (__bf16*)take((size_t)1536 * 768 * 2);
    __bf16* wbDiff2 = (__bf16*)take((size_t)768 * 384 * 2);
    __bf16* act1    = (__bf16*)take((size_t)B_ROWS * 768 * 2);
    __bf16* act2    = (__bf16*)take((size_t)B_ROWS * 384 * 2);
    float*  encbuf  = (float*)take((size_t)B_ROWS * 64 * 4);
    float*  norms   = (float*)take((size_t)4 * B_ROWS * 4);

    const dim3 blk(256);
    auto pack = [&](const float* s, __bf16* d, int K, int N) {
        const int total = (K >> 5) * N * 2;
        pack_w_bf16<<<dim3((total + 255) / 256), blk, 0, stream>>>(s, d, K, N);
    };
    pack(info_w1, wbInfo1, 768, 256);
    pack(info_w2, wbInfo2, 256, 64);
    pack(imp_w1,  wbImp1,  768, 384);
    pack(disc_w1, wbDisc1, 768, 256);
    pack(disc_w2, wbDisc2, 256, 64);
    pack(cons_w1, wbCons1, 1536, 768);
    pack(cons_w2, wbCons2, 768, 384);
    pack(diff_w1, wbDiff1, 1536, 768);
    pack(diff_w2, wbDiff2, 768, 384);

    // ---- geometric quality: rows 0..3 + saved norms ----
    for (int i = 0; i < 4; ++i)
        geometric_kernel<<<dim3(B_ROWS / 8), blk, 0, stream>>>(
            feats[i], out + (size_t)i * B_ROWS, norms + (size_t)i * B_ROWS);

    auto gemm_f32 = [&](const float* A, int K, int N, const __bf16* W,
                        const float* bias, __bf16* o) {
        gemm_bias_act<true, false, true, true>
            <<<dim3(B_ROWS / 128, N / 64), blk, 0, stream>>>(A, A, K, K, K, W, bias, o, N);
    };
    auto gemm_bf16 = [&](const __bf16* A, int K, int N, const __bf16* W,
                         const float* bias, __bf16* o) {
        gemm_bias_act<false, false, true, true>
            <<<dim3(B_ROWS / 128, N / 64), blk, 0, stream>>>(A, A, K, K, K, W, bias, o, N);
    };
    auto gemm_concat = [&](const float* A0, const float* A1, const __bf16* W,
                           const float* bias, __bf16* o) {
        gemm_bias_act<true, true, true, true>
            <<<dim3(B_ROWS / 128, 768 / 64), blk, 0, stream>>>(A0, A1, 768, 768, 1536, W, bias, o, 768);
    };
    auto dot_sig = [&](const __bf16* A, int Kd, const float* w, const float* b, float* o) {
        dot_sigmoid_kernel<<<dim3(B_ROWS / 8), blk, 0, stream>>>(A, Kd, w, b, o);
    };

    // ---- information quality: rows 4,5 (image, text) ----
    for (int i = 0; i < 2; ++i) {
        gemm_f32(feats[i], 768, 256, wbInfo1, info_b1, act1);
        gemm_bias_act<false, false, false, false>
            <<<dim3(B_ROWS / 128, 1), blk, 0, stream>>>(
                act1, act1, 256, 256, 256, wbInfo2, info_b2, encbuf, 64);
        info_finish_kernel<<<dim3(B_ROWS / 256), blk, 0, stream>>>(
            encbuf, norms + (size_t)i * B_ROWS, out + (size_t)(4 + i) * B_ROWS);
    }

    // ---- modality importance: rows 8..11 (img, txt, eimg, etxt) ----
    for (int i = 0; i < 4; ++i) {
        gemm_f32(feats[i], 768, 384, wbImp1, imp_b1, act2);
        dot_sig(act2, 384, imp_w2, imp_b2, out + (size_t)(8 + i) * B_ROWS);
    }

    // ---- feature discriminator: rows 12..15 = img, eimg, txt, etxt ----
    {
        const int fi[4] = { 0, 2, 1, 3 };
        for (int j = 0; j < 4; ++j) {
            gemm_f32(feats[fi[j]], 768, 256, wbDisc1, disc_b1, act1);
            gemm_bf16(act1, 256, 64, wbDisc2, disc_b2, act2);
            dot_sig(act2, 64, disc_w3, disc_b3, out + (size_t)(12 + j) * B_ROWS);
        }
    }

    // ---- consistency: row 6 (orig pair), row 7 (enhanced pair) ----
    for (int j = 0; j < 2; ++j) {
        const float* A0 = (j == 0) ? feats[0] : feats[2];
        const float* A1 = (j == 0) ? feats[1] : feats[3];
        gemm_concat(A0, A1, wbCons1, cons_b1, act1);
        gemm_bf16(act1, 768, 384, wbCons2, cons_b2, act2);
        dot_sig(act2, 384, cons_w3, cons_b3, out + (size_t)(6 + j) * B_ROWS);
    }

    // ---- task difficulty: row 16 (enhanced pair) ----
    gemm_concat(feats[2], feats[3], wbDiff1, diff_b1, act1);
    gemm_bf16(act1, 768, 384, wbDiff2, diff_b2, act2);
    dot_sig(act2, 384, diff_w3, diff_b3, out + (size_t)16 * B_ROWS);

    // ---- overall: row 17 ----
    overall_kernel<<<dim3(B_ROWS / 256), blk, 0, stream>>>(out);
}
